// GCNLayer_46583215292666
// MI455X (gfx1250) — matrix operand order repaired and checked by reading, not compiled, not run on hardware
//
#include <hip/hip_runtime.h>
#include <hip/hip_bf16.h>

typedef __attribute__((ext_vector_type(16))) __bf16 v16bf;
typedef __attribute__((ext_vector_type(8)))  float  v8f;
typedef __attribute__((ext_vector_type(4)))  float  v4f;

// ---------------------------------------------------------------------------
// Shapes: B=4, N=256, K=16, D=128.  Edge rows = B*N*N = 262144, flat (rows, 128).
// ---------------------------------------------------------------------------

__device__ __forceinline__ v8f v8f_zero() {
    v8f z;
#pragma unroll
    for (int i = 0; i < 8; ++i) z[i] = 0.f;
    return z;
}

// Sum across the 16-lane half-group (xor masks 1,2,4,8 stay within each half).
__device__ __forceinline__ float hsum16(float v) {
    v += __shfl_xor(v, 1, 32);
    v += __shfl_xor(v, 2, 32);
    v += __shfl_xor(v, 4, 32);
    v += __shfl_xor(v, 8, 32);
    return v;
}

// Build 4 A-operand chunks (16x32 bf16 each) covering K=0..127 from a 16x128
// fp32 tile in LDS.  ISA A layout (16-bit, 16x32):
//   lane<16 : halves 0-7 -> K=0..7,  halves 8-15 -> K=16..23
//   lane>=16: halves 0-7 -> K=8..15, halves 8-15 -> K=24..31   (M = lane&15)
__device__ __forceinline__ void build_a(v16bf A[4], const float* ldsE, int l15, int hs) {
#pragma unroll
    for (int kc = 0; kc < 4; ++kc) {
        const float* row = ldsE + l15 * 128 + kc * 32 + hs * 8;
        v16bf a;
#pragma unroll
        for (int h = 0; h < 8; ++h) a[h] = (__bf16)row[h];
#pragma unroll
        for (int h = 0; h < 8; ++h) a[8 + h] = (__bf16)row[16 + h];
        A[kc] = a;
    }
}

// acc(16x128) += A(16x128) * Wmat(128x128 in pre-swizzled B layout)
// B-operand block (kc,nt) is 512 bf16: lane*16 halves, half h -> K = kc*32 + (lane<16?0:8) + (h<8?h:8+h),
// n = nt*16 + (lane&15).
__device__ __forceinline__ void gemm_acc(v8f acc[8], const v16bf A[4],
                                         const __bf16* wmat, int lane) {
#pragma unroll
    for (int kc = 0; kc < 4; ++kc) {
#pragma unroll
        for (int nt = 0; nt < 8; ++nt) {
            v16bf bop = *(const v16bf*)(wmat + (kc * 8 + nt) * 512 + lane * 16);
            acc[nt] = __builtin_amdgcn_wmma_f32_16x16x32_bf16(
                false, A[kc], false, bop, (short)0, acc[nt], false, false);
        }
    }
}

// ---------------------------------------------------------------------------
// Precompute: Vc = Va @ V_edge_in (Va = V_edge[:, :128]),  bc = Va@b_vein + b_V_edge
// ---------------------------------------------------------------------------
__global__ void prep_vc(const float* __restrict__ Ve, const float* __restrict__ Vin,
                        const float* __restrict__ bVin, const float* __restrict__ bVe,
                        float* __restrict__ Vc, float* __restrict__ bc) {
    __shared__ float va[128];
    const int o = blockIdx.x, i = threadIdx.x;
    va[i] = Ve[(size_t)o * 256 + i];
    __syncthreads();
    float s = 0.f;
    for (int j = 0; j < 128; ++j) s += va[j] * Vin[j * 128 + i];
    Vc[o * 128 + i] = s;
    if (i == 0) {
        float t = bVe[o];
        for (int j = 0; j < 128; ++j) t += va[j] * bVin[j];
        bc[o] = t;
    }
}

// ---------------------------------------------------------------------------
// Convert the 4 edge weight matrices to bf16 in WMMA B-operand swizzled layout.
// mats: 0=W1_edge, 1=W_edge, 2=Vc, 3=Vb(=V_edge[:,128:256]).  B[k][n] = W[n][k].
// ---------------------------------------------------------------------------
__global__ void prep_w(const float* __restrict__ W1, const float* __restrict__ We,
                       const float* __restrict__ Vc, const float* __restrict__ Ve,
                       __bf16* __restrict__ prep) {
    const int gid = blockIdx.x * blockDim.x + threadIdx.x;  // 0..65535
    const int mat = gid >> 14;
    const int p   = gid & 16383;
    const int blk = p >> 9;
    const int kc  = blk >> 3, nt = blk & 7;
    const int qq  = p & 511;
    const int l   = qq >> 4, h = qq & 15;
    const int n   = nt * 16 + (l & 15);
    const int K   = kc * 32 + ((l < 16) ? 0 : 8) + ((h < 8) ? h : 8 + h);
    float v;
    if      (mat == 0) v = W1[n * 128 + K];
    else if (mat == 1) v = We[n * 128 + K];
    else if (mat == 2) v = Vc[n * 128 + K];
    else               v = Ve[n * 256 + 128 + K];
    prep[gid] = (__bf16)v;
}

// ---------------------------------------------------------------------------
// Node precompute per (b,n): q, t_n = x@V_node_in^T+b, r2 = x@W2^T + (bW1+bW2+bW3),
// r3 = x@W3^T.
// ---------------------------------------------------------------------------
__global__ void node_pre(const float* __restrict__ x,
                         const float* __restrict__ Wq, const float* __restrict__ bq,
                         const float* __restrict__ Vni, const float* __restrict__ bVni,
                         const float* __restrict__ W2, const float* __restrict__ W3,
                         const float* __restrict__ bW1, const float* __restrict__ bW2,
                         const float* __restrict__ bW3,
                         float* __restrict__ q, float* __restrict__ tn,
                         float* __restrict__ r2, float* __restrict__ r3) {
    __shared__ float xr[128];
    const int row = blockIdx.x, o = threadIdx.x;
    xr[o] = x[(size_t)row * 128 + o];
    __syncthreads();
    float aq = 0.f, at = 0.f, a2 = 0.f, a3 = 0.f;
    for (int i = 0; i < 128; ++i) {
        const float xv = xr[i];
        aq += xv * Wq[o * 128 + i];
        at += xv * Vni[o * 128 + i];
        a2 += xv * W2[o * 128 + i];
        a3 += xv * W3[o * 128 + i];
    }
    q[(size_t)row * 128 + o]  = aq + bq[o];
    tn[(size_t)row * 128 + o] = at + bVni[o];
    r2[(size_t)row * 128 + o] = a2 + bW1[o] + bW2[o] + bW3[o];
    r3[(size_t)row * 128 + o] = a3;
}

// k,v for gathered neighbor rows (B*N*K = 16384 rows)
__global__ void node_kv(const float* __restrict__ x, const int* __restrict__ idx,
                        const float* __restrict__ Wk, const float* __restrict__ bk,
                        const float* __restrict__ Wv, const float* __restrict__ bv,
                        float* __restrict__ kk, float* __restrict__ vv) {
    __shared__ float xr[128];
    const int r = blockIdx.x;
    const int b = r >> 12;  // /(N*K)
    const int src = idx[r];
    const int o = threadIdx.x;
    xr[o] = x[((size_t)b * 256 + src) * 128 + o];
    __syncthreads();
    float ak = 0.f, av = 0.f;
    for (int i = 0; i < 128; ++i) {
        const float xv = xr[i];
        ak += xv * Wk[o * 128 + i];
        av += xv * Wv[o * 128 + i];
    }
    kk[(size_t)r * 128 + o] = ak + bk[o];
    vv[(size_t)r * 128 + o] = av + bv[o];
}

__device__ __forceinline__ void block_ln_stats(float v, float* red, float* mu, float* rs) {
    const int o = threadIdx.x;
    red[o] = v;
    __syncthreads();
    for (int s = 64; s > 0; s >>= 1) { if (o < s) red[o] += red[o + s]; __syncthreads(); }
    const float sum = red[0];
    __syncthreads();
    red[o] = v * v;
    __syncthreads();
    for (int s = 64; s > 0; s >>= 1) { if (o < s) red[o] += red[o + s]; __syncthreads(); }
    const float sq = red[0];
    __syncthreads();
    const float m = sum * (1.f / 128.f);
    *mu = m;
    *rs = rsqrtf(sq * (1.f / 128.f) - m * m + 1e-5f);
}

// attention + node FFN, one block per (b,n)
__global__ void node_main(const float* __restrict__ x, const float* __restrict__ q,
                          const float* __restrict__ kmat, const float* __restrict__ vmat,
                          const float* __restrict__ tn,
                          const float* __restrict__ Wn, const float* __restrict__ bWn,
                          const float* __restrict__ Vn, const float* __restrict__ bVn,
                          const float* __restrict__ g1, const float* __restrict__ be1,
                          const float* __restrict__ g2, const float* __restrict__ be2,
                          float* __restrict__ out) {
    __shared__ float qs[128], sc[16], attn[128], hnb[128], red[128], xr[128], tns[128];
    const int row = blockIdx.x, o = threadIdx.x;
    qs[o]  = q[(size_t)row * 128 + o];
    xr[o]  = x[(size_t)row * 128 + o];
    tns[o] = tn[(size_t)row * 128 + o];
    __syncthreads();
    if (o < 16) {
        const float* kr = kmat + ((size_t)row * 16 + o) * 128;
        float s = 0.f;
        for (int i = 0; i < 128; ++i) s += qs[i] * kr[i];
        sc[o] = s * 0.08838834764831845f;  // 1/sqrt(128)
    }
    __syncthreads();
    if (o == 0) {
        float mx = sc[0];
        for (int t = 1; t < 16; ++t) mx = fmaxf(mx, sc[t]);
        float sum = 0.f;
        for (int t = 0; t < 16; ++t) { const float ev = __expf(sc[t] - mx); sc[t] = ev; sum += ev; }
        const float inv = 1.f / sum;
        for (int t = 0; t < 16; ++t) sc[t] *= inv;
    }
    __syncthreads();
    {
        float a = 0.f;
        for (int t = 0; t < 16; ++t) a += sc[t] * vmat[((size_t)row * 16 + t) * 128 + o];
        attn[o] = a;
    }
    __syncthreads();
    float y = 0.f;
    for (int i = 0; i < 128; ++i) y += attn[i] * Wn[o * 128 + i];
    y = fmaxf(y + bWn[o], 0.f);
    const float z = xr[o] + y;
    float mu, rs;
    block_ln_stats(z, red, &mu, &rs);
    const float h = (z - mu) * rs * g1[o] + be1[o];
    hnb[o] = h;
    __syncthreads();
    float y2 = 0.f;
    for (int i = 0; i < 128; ++i) y2 += tns[i] * Vn[o * 256 + i];
    for (int i = 0; i < 128; ++i) y2 += hnb[i] * Vn[o * 256 + 128 + i];
    y2 = fmaxf(y2 + bVn[o], 0.f);
    const float z2 = h + y2;
    float mu2, rs2;
    block_ln_stats(z2, red, &mu2, &rs2);
    out[(size_t)row * 128 + o] = (z2 - mu2) * rs2 * g2[o] + be2[o];
}

// ---------------------------------------------------------------------------
// Fused edge kernel.  4 waves/WG, each wave owns a 16-row tile of (262144,128).
// LDS: 128KB bf16 weights (W1, W_edge, Vc, Vb) + 8KB fp32 scratch per wave.
// Staging uses gfx1250 async global->LDS DMA (ASYNCcnt) — no VGPR roundtrip.
// ---------------------------------------------------------------------------
__launch_bounds__(128)
__global__ void edge_fused(const float* __restrict__ e,
                           const float* __restrict__ r2, const float* __restrict__ r3,
                           const __bf16* __restrict__ prep,
                           const float* __restrict__ bWe, const float* __restrict__ bc,
                           const float* __restrict__ g1, const float* __restrict__ be1,
                           const float* __restrict__ g2, const float* __restrict__ be2,
                           float* __restrict__ out) {
    extern __shared__ char smem[];
    __bf16* w = (__bf16*)smem;                       // 4 * 16384 halves = 128KB
    float*  ldsbase = (float*)(smem + 4 * 16384 * 2);
    const int tid = threadIdx.x;

    const int wave = tid >> 5;
    const int lane = tid & 31;
    const int hs   = lane >> 4;   // half-group select
    const int l15  = lane & 15;
    float* ldsE = ldsbase + wave * 2048;  // 16x128 fp32 scratch

    const int r0 = (blockIdx.x * 4 + wave) * 16;  // first flat edge row of tile
    const int b  = r0 >> 16;
    const int n  = (r0 >> 8) & 255;
    const int m0 = r0 & 255;
    const float* esrc = e + (size_t)r0 * 128;

    // --- async DMA: swizzled weights (whole WG, 128KB) straight into LDS ---
    // extern __shared__ block starts at LDS byte offset 0 (no static LDS here).
    {
        const uint4* gsrc = (const uint4*)prep;
        for (int i = tid; i < 8192; i += 128) {
            unsigned la = (unsigned)i * 16u;
            asm volatile("global_load_async_to_lds_b128 %0, %1, off"
                         :: "v"(la), "v"(gsrc + i) : "memory");
        }
    }
    // --- async DMA: this wave's 16x128 fp32 e tile into its scratch region ---
    {
        const unsigned ebase = 131072u + (unsigned)wave * 8192u + (unsigned)lane * 16u;
#pragma unroll
        for (int i = 0; i < 16; ++i) {
            unsigned la = ebase + (unsigned)i * 512u;
            asm volatile("global_load_async_to_lds_b128 %0, %1, off"
                         :: "v"(la), "v"((const v4f*)esrc + (i * 32 + lane)) : "memory");
        }
    }
    asm volatile("s_wait_asynccnt 0x0" ::: "memory");
    __syncthreads();  // weights visible across all 4 waves

    v16bf eA[4];
    build_a(eA, ldsE, l15, hs);

    // per-lane column vectors (n-indexed), reused across all j
    float bWev[8], g1v[8], be1v[8], g2v[8], be2v[8], bcv[8];
#pragma unroll
    for (int nt = 0; nt < 8; ++nt) {
        const int ng = nt * 16 + l15;
        bWev[nt] = bWe[ng]; bcv[nt] = bc[ng];
        g1v[nt] = g1[ng]; be1v[nt] = be1[ng];
        g2v[nt] = g2[ng]; be2v[nt] = be2[ng];
    }

    // ---- GEMM1: pre = e @ W1^T + r2[b,n] + r3[b,m] (+ folded biases), relu ----
    v8f acc1[8];
#pragma unroll
    for (int i = 0; i < 8; ++i) acc1[i] = v8f_zero();
    gemm_acc(acc1, eA, w, lane);

    const float* r2row = r2 + ((size_t)(b * 256 + n)) * 128;
    const float* r3b   = r3 + ((size_t)b * 256) * 128;
#pragma unroll
    for (int nt = 0; nt < 8; ++nt) {
        const int ng = nt * 16 + l15;
        const float r2v = r2row[ng];
#pragma unroll
        for (int j = 0; j < 8; ++j) {
            const int m = m0 + j + 8 * hs;
            const float v = acc1[nt][j] + r2v + r3b[m * 128 + ng];
            ldsE[(j + 8 * hs) * 128 + ng] = fmaxf(v, 0.f);  // overwrite e with relu(pre)
        }
    }

    v16bf pA[4];
    build_a(pA, ldsE, l15, hs);

    // ---- GEMM2: relu(pre) @ W_edge^T + bWe, relu, + e residual, LN1 -> h1 ----
    v8f acc2[8];
#pragma unroll
    for (int i = 0; i < 8; ++i) acc2[i] = v8f_zero();
    gemm_acc(acc2, pA, w + 16384, lane);

#pragma unroll
    for (int nt = 0; nt < 8; ++nt) {
        const int ng = nt * 16 + l15;
#pragma unroll
        for (int j = 0; j < 8; ++j) {
            const float v = fmaxf(acc2[nt][j] + bWev[nt], 0.f);
            acc2[nt][j] = v + esrc[(size_t)(j + 8 * hs) * 128 + ng];  // e is L2-resident
        }
    }
#pragma unroll
    for (int j = 0; j < 8; ++j) {
        float s = 0.f, s2 = 0.f;
#pragma unroll
        for (int nt = 0; nt < 8; ++nt) { const float v = acc2[nt][j]; s += v; s2 += v * v; }
        s = hsum16(s); s2 = hsum16(s2);
        const float mu = s * (1.f / 128.f);
        const float rs = rsqrtf(s2 * (1.f / 128.f) - mu * mu + 1e-5f);
#pragma unroll
        for (int nt = 0; nt < 8; ++nt)
            acc2[nt][j] = (acc2[nt][j] - mu) * rs * g1v[nt] + be1v[nt];
    }

    // h1 (acc2) -> LDS -> A layout
#pragma unroll
    for (int nt = 0; nt < 8; ++nt) {
        const int ng = nt * 16 + l15;
#pragma unroll
        for (int j = 0; j < 8; ++j)
            ldsE[(j + 8 * hs) * 128 + ng] = acc2[nt][j];
    }
    build_a(pA, ldsE, l15, hs);  // reuse pA as h1A

    // ---- GEMM3: y = e@Vc^T + h1@Vb^T + bc, relu, + h1, LN2, store ----
#pragma unroll
    for (int i = 0; i < 8; ++i) acc1[i] = v8f_zero();
    gemm_acc(acc1, eA, w + 2 * 16384, lane);
    gemm_acc(acc1, pA, w + 3 * 16384, lane);

#pragma unroll
    for (int nt = 0; nt < 8; ++nt) {
#pragma unroll
        for (int j = 0; j < 8; ++j)
            acc1[nt][j] = acc2[nt][j] + fmaxf(acc1[nt][j] + bcv[nt], 0.f);
    }
#pragma unroll
    for (int j = 0; j < 8; ++j) {
        float s = 0.f, s2 = 0.f;
#pragma unroll
        for (int nt = 0; nt < 8; ++nt) { const float v = acc1[nt][j]; s += v; s2 += v * v; }
        s = hsum16(s); s2 = hsum16(s2);
        const float mu = s * (1.f / 128.f);
        const float rs = rsqrtf(s2 * (1.f / 128.f) - mu * mu + 1e-5f);
#pragma unroll
        for (int nt = 0; nt < 8; ++nt)
            acc1[nt][j] = (acc1[nt][j] - mu) * rs * g2v[nt] + be2v[nt];
    }

    float* oedge = out + 131072 + (size_t)r0 * 128;  // h_edge follows h_node
#pragma unroll
    for (int nt = 0; nt < 8; ++nt) {
        const int ng = nt * 16 + l15;
#pragma unroll
        for (int j = 0; j < 8; ++j)
            oedge[(size_t)(j + 8 * hs) * 128 + ng] = acc1[nt][j];
    }
}

// ---------------------------------------------------------------------------
extern "C" void kernel_launch(void* const* d_in, const int* in_sizes, int n_in,
                              void* d_out, int out_size, void* d_ws, size_t ws_size,
                              hipStream_t stream) {
    const float* x      = (const float*)d_in[0];
    const float* e      = (const float*)d_in[1];
    const int*   nidx   = (const int*)d_in[2];
    const float* Wq     = (const float*)d_in[3];
    const float* bq     = (const float*)d_in[4];
    const float* Wk     = (const float*)d_in[5];
    const float* bk     = (const float*)d_in[6];
    const float* Wv     = (const float*)d_in[7];
    const float* bv     = (const float*)d_in[8];
    const float* W_node = (const float*)d_in[9];
    const float* b_Wn   = (const float*)d_in[10];
    const float* Vni    = (const float*)d_in[11];
    const float* b_Vni  = (const float*)d_in[12];
    const float* V_node = (const float*)d_in[13];
    const float* b_Vn   = (const float*)d_in[14];
    const float* g1n    = (const float*)d_in[15];
    const float* be1n   = (const float*)d_in[16];
    const float* g2n    = (const float*)d_in[17];
    const float* be2n   = (const float*)d_in[18];
    const float* W_edge = (const float*)d_in[19];
    const float* b_We   = (const float*)d_in[20];
    const float* Vei    = (const float*)d_in[21];
    const float* b_Vei  = (const float*)d_in[22];
    const float* V_edge = (const float*)d_in[23];
    const float* b_Ve   = (const float*)d_in[24];
    const float* W1e    = (const float*)d_in[25];
    const float* b_W1e  = (const float*)d_in[26];
    const float* W2e    = (const float*)d_in[27];
    const float* b_W2e  = (const float*)d_in[28];
    const float* W3e    = (const float*)d_in[29];
    const float* b_W3e  = (const float*)d_in[30];
    const float* g1e    = (const float*)d_in[31];
    const float* be1e   = (const float*)d_in[32];
    const float* g2e    = (const float*)d_in[33];
    const float* be2e   = (const float*)d_in[34];

    char* ws = (char*)d_ws;
    float* ws_q  = (float*)ws; ws += 131072 * 4;
    float* ws_tn = (float*)ws; ws += 131072 * 4;
    float* ws_r2 = (float*)ws; ws += 131072 * 4;
    float* ws_r3 = (float*)ws; ws += 131072 * 4;
    float* ws_k  = (float*)ws; ws += 2097152 * 4;
    float* ws_v  = (float*)ws; ws += 2097152 * 4;
    float* ws_Vc = (float*)ws; ws += 16384 * 4;
    float* ws_bc = (float*)ws; ws += 512;
    __bf16* ws_prep = (__bf16*)ws; ws += 65536 * 2;

    prep_vc<<<128, 128, 0, stream>>>(V_edge, Vei, b_Vei, b_Ve, ws_Vc, ws_bc);
    prep_w<<<256, 256, 0, stream>>>(W1e, W_edge, ws_Vc, V_edge, ws_prep);
    node_pre<<<1024, 128, 0, stream>>>(x, Wq, bq, Vni, b_Vni, W2e, W3e,
                                       b_W1e, b_W2e, b_W3e, ws_q, ws_tn, ws_r2, ws_r3);
    node_kv<<<16384, 128, 0, stream>>>(x, nidx, Wk, bk, Wv, bv, ws_k, ws_v);
    node_main<<<1024, 128, 0, stream>>>(x, ws_q, ws_k, ws_v, ws_tn, W_node, b_Wn,
                                        V_node, b_Vn, g1n, be1n, g2n, be2n, (float*)d_out);

    const size_t shmem = 4 * 16384 * 2 + 4 * 2048 * 4;  // 128KB weights + 32KB scratch
    edge_fused<<<4096, 128, shmem, stream>>>(e, ws_r2, ws_r3, ws_prep, b_We, ws_bc,
                                             g1e, be1e, g2e, be2e, (float*)d_out);
}
